// TrajCls_53996328845485
// MI455X (gfx1250) — matrix operand order repaired and checked by reading, not compiled, not run on hardware
//
#include <hip/hip_runtime.h>
#include <math.h>

// ---------------- model dimensions ----------------
#define Bz   64
#define Sz   256
#define Dz   128
#define Hz   8
#define DHz  16
#define FFz  512
#define Lz   4
#define Uz   5000
#define UPz  5056          // head N padded to a multiple of 64
#define Nn   40000
#define Ez   160000
#define FEAz 32
#define CEDz 16
#define TDz  6
#define FUSEK 22
#define FUSEKP 32          // fusion K padded to 32
#define BS   (Bz * Sz)     // 16384

typedef __attribute__((ext_vector_type(16))) _Float16 v16h;
typedef __attribute__((ext_vector_type(8)))  _Float16 v8h;
typedef __attribute__((ext_vector_type(8)))  float    v8f;

// =====================================================================
// WMMA GEMM, fully tiled (M%32==0, N%64==0, K%32==0 — guaranteed by
// caller-side padding).  256 threads = 8 waves, block tile 32x64.
// A (32x32) and W (32x64, stored transposed) staged in LDS as f16 with
// float4 global loads; fragments are contiguous ds_load_b128 reads.
// =====================================================================
__global__ __launch_bounds__(256)
void gemm_wmma_f16(const float* __restrict__ A,
                   const float* __restrict__ W,
                   const float* __restrict__ bias,
                   const float* __restrict__ residual,
                   float* __restrict__ C,
                   int M, int N, int K, int relu)
{
    __shared__ __align__(16) _Float16 As[32 * 32];   // [row][k]
    __shared__ __align__(16) _Float16 Wt[64 * 32];   // [col][k]

    const int lane = threadIdx.x & 31;
    const int w    = threadIdx.x >> 5;
    const int hi   = lane >> 4;
    const int m16  = lane & 15;

    const int bm = blockIdx.y * 32;
    const int bn = blockIdx.x * 64;
    const int rowL = (w >> 2) * 16 + m16;   // wave-local A row
    const int colL = (w & 3) * 16 + m16;    // wave-local W col

    // staging indices
    const int ar = threadIdx.x >> 3;          // 0..31  A row
    const int aq = (threadIdx.x & 7) * 4;     // 0..28  A k quad
    const float* srcA = A + (size_t)(bm + ar) * K + aq;

    v8f acc = {};
    for (int k0 = 0; k0 < K; k0 += 32) {
        // ---- stage A slab: one float4 per thread ----
        {
            if (k0 + 32 < K) __builtin_prefetch(srcA + k0 + 32, 0, 1);
            const float4 v = *(const float4*)(srcA + k0);
            _Float16* d = &As[ar * 32 + aq];
            d[0] = (_Float16)v.x; d[1] = (_Float16)v.y;
            d[2] = (_Float16)v.z; d[3] = (_Float16)v.w;
        }
        // ---- stage W slab transposed: two float4 per thread ----
        #pragma unroll
        for (int i = 0; i < 2; ++i) {
            int idx = threadIdx.x + i * 256;        // 0..511
            int kr  = idx >> 4;                     // 0..31
            int ci  = (idx & 15) * 4;               // 0..60
            const float4 v = *(const float4*)(W + (size_t)(k0 + kr) * N + bn + ci);
            Wt[(ci + 0) * 32 + kr] = (_Float16)v.x;
            Wt[(ci + 1) * 32 + kr] = (_Float16)v.y;
            Wt[(ci + 2) * 32 + kr] = (_Float16)v.z;
            Wt[(ci + 3) * 32 + kr] = (_Float16)v.w;
        }
        __syncthreads();

        // ---- fragments: contiguous 16B LDS reads ----
        const v8h a0 = *(const v8h*)&As[rowL * 32 + hi * 8];
        const v8h a1 = *(const v8h*)&As[rowL * 32 + 16 + hi * 8];
        const v8h b0 = *(const v8h*)&Wt[colL * 32 + hi * 8];
        const v8h b1 = *(const v8h*)&Wt[colL * 32 + 16 + hi * 8];
        v16h a, b;
        #pragma unroll
        for (int j = 0; j < 8; ++j) {
            a[j] = a0[j]; a[j + 8] = a1[j];
            b[j] = b0[j]; b[j + 8] = b1[j];
        }
        acc = __builtin_amdgcn_wmma_f32_16x16x32_f16(false, a, false, b,
                                                     (short)0, acc, false, false);
        __syncthreads();
    }

    #pragma unroll
    for (int r = 0; r < 8; ++r) {
        int row = bm + (w >> 2) * 16 + r + hi * 8;
        int col = bn + colL;
        float v = acc[r];
        if (bias)     v += bias[col];
        if (residual) v += residual[(size_t)row * N + col];
        if (relu)     v = fmaxf(v, 0.0f);
        C[(size_t)row * N + col] = v;
    }
}

// =====================================================================
// Precompute attention bias:  bias[b,q,k] = mask - lam*log1p(T) -
// (1-lam)*log1p(Dist).  Reads the T/D matrices (16MB each) exactly once.
// =====================================================================
__global__ void bias_pre_kernel(const float* __restrict__ tmat,
                                const float* __restrict__ dmat,
                                const int*   __restrict__ mask,
                                const float* __restrict__ lam_p,
                                float* __restrict__ biasb)
{
    int idx = blockIdx.x * blockDim.x + threadIdx.x;
    if (idx >= Bz * Sz * Sz) return;
    int b  = idx / (Sz * Sz);
    int kk = idx % Sz;
    float lam = *lam_p;
    float rel = lam * logf(1.0f + tmat[idx]) + (1.0f - lam) * logf(1.0f + dmat[idx]);
    float mb  = (mask[b * Sz + kk] > 0) ? 0.0f : -1e9f;
    biasb[idx] = mb - rel;
}

// =====================================================================
// Fused attention for one (b,h).  K row-major and V transposed staged
// in LDS (f16); per-wave 16x256 f32 score strip in LDS; bias folded
// into the softmax pass (contiguous reads).
// =====================================================================
__global__ __launch_bounds__(256)
void attn_wmma(const float* __restrict__ qkv,    // [B,S,3D]
               const float* __restrict__ biasb,  // [B,S,S]
               float* __restrict__ out)          // [B,S,D]
{
    extern __shared__ char smem[];
    _Float16* Ksh = (_Float16*)smem;              // [256][16] row-major
    _Float16* Vt  = Ksh + Sz * DHz;               // [16][256] transposed
    float*    sc  = (float*)(smem + 2 * Sz * DHz * sizeof(_Float16)); // [8][16][256]

    const int b = blockIdx.x / Hz;
    const int h = blockIdx.x % Hz;
    const int lane = threadIdx.x & 31;
    const int w    = threadIdx.x >> 5;
    const int hi   = lane >> 4;
    const int m16  = lane & 15;
    const float scale = 0.25f;   // 1/sqrt(DH)

    // ---- stage K (row-major) and V (transposed), float4 loads ----
    {
        int s = threadIdx.x;
        const float* row = qkv + (size_t)(b * Sz + s) * (3 * Dz) + h * DHz;
        #pragma unroll
        for (int q4 = 0; q4 < 4; ++q4) {
            const float4 kv = *(const float4*)(row + Dz + q4 * 4);
            const float4 vv = *(const float4*)(row + 2 * Dz + q4 * 4);
            _Float16* kd = &Ksh[s * DHz + q4 * 4];
            kd[0] = (_Float16)kv.x; kd[1] = (_Float16)kv.y;
            kd[2] = (_Float16)kv.z; kd[3] = (_Float16)kv.w;
            Vt[(q4 * 4 + 0) * Sz + s] = (_Float16)vv.x;
            Vt[(q4 * 4 + 1) * Sz + s] = (_Float16)vv.y;
            Vt[(q4 * 4 + 2) * Sz + s] = (_Float16)vv.z;
            Vt[(q4 * 4 + 3) * Sz + s] = (_Float16)vv.w;
        }
    }
    __syncthreads();

    float* scw = sc + (size_t)w * 16 * Sz;

    for (int strip = 0; strip < 2; ++strip) {
        const int qbase = strip * 128 + w * 16;

        // ---- Q fragment: d padded 16->32 with zeros, scale folded in
        v16h qa = {};
        {
            int q = qbase + m16;
            const float* qrow = qkv + (size_t)(b * Sz + q) * (3 * Dz) + h * DHz;
            const float4 q0 = *(const float4*)(qrow + hi * 8);
            const float4 q1 = *(const float4*)(qrow + hi * 8 + 4);
            qa[0] = (_Float16)(q0.x * scale); qa[1] = (_Float16)(q0.y * scale);
            qa[2] = (_Float16)(q0.z * scale); qa[3] = (_Float16)(q0.w * scale);
            qa[4] = (_Float16)(q1.x * scale); qa[5] = (_Float16)(q1.y * scale);
            qa[6] = (_Float16)(q1.z * scale); qa[7] = (_Float16)(q1.w * scale);
        }

        // ---- scores: 16 key tiles of 16
        for (int t = 0; t < 16; ++t) {
            int kk = t * 16 + m16;
            v16h kbf = {};
            {   // d = 8*hi + e for e < 8; elements 8..15 carry the zero padding
                const v8h b0 = *(const v8h*)&Ksh[kk * DHz + hi * 8];
                #pragma unroll
                for (int j = 0; j < 8; ++j) { kbf[j] = b0[j]; }
            }
            v8f cs = {};
            cs = __builtin_amdgcn_wmma_f32_16x16x32_f16(false, qa, false, kbf,
                                                        (short)0, cs, false, false);
            #pragma unroll
            for (int r = 0; r < 8; ++r)
                scw[(r + hi * 8) * Sz + kk] = cs[r];
        }

        // ---- bias add + softmax (lane r owns row r, contiguous reads)
        if (lane < 16) {
            int q = qbase + lane;
            const float* br = biasb + (size_t)(b * Sz + q) * Sz;
            float* row = scw + lane * Sz;
            float mx = -INFINITY;
            for (int j = 0; j < Sz; ++j) {
                float v = row[j] + br[j];
                row[j] = v;
                mx = fmaxf(mx, v);
            }
            float sum = 0.0f;
            for (int j = 0; j < Sz; ++j) { float e = expf(row[j] - mx); row[j] = e; sum += e; }
            float inv = 1.0f / sum;
            for (int j = 0; j < Sz; ++j) row[j] *= inv;
        }

        // ---- O = P @ V : 8 chunks of 32 keys
        v8f oacc = {};
        for (int c8 = 0; c8 < 8; ++c8) {
            const float* prow = scw + m16 * Sz + c8 * 32;
            const float4 p0 = *(const float4*)(prow + hi * 8);
            const float4 p1 = *(const float4*)(prow + hi * 8 + 4);
            const float4 p2 = *(const float4*)(prow + 16 + hi * 8);
            const float4 p3 = *(const float4*)(prow + 16 + hi * 8 + 4);
            v16h pa;
            pa[0]  = (_Float16)p0.x; pa[1]  = (_Float16)p0.y;
            pa[2]  = (_Float16)p0.z; pa[3]  = (_Float16)p0.w;
            pa[4]  = (_Float16)p1.x; pa[5]  = (_Float16)p1.y;
            pa[6]  = (_Float16)p1.z; pa[7]  = (_Float16)p1.w;
            pa[8]  = (_Float16)p2.x; pa[9]  = (_Float16)p2.y;
            pa[10] = (_Float16)p2.z; pa[11] = (_Float16)p2.w;
            pa[12] = (_Float16)p3.x; pa[13] = (_Float16)p3.y;
            pa[14] = (_Float16)p3.z; pa[15] = (_Float16)p3.w;

            const v8h v0 = *(const v8h*)&Vt[m16 * Sz + c8 * 32 + hi * 8];
            const v8h v1 = *(const v8h*)&Vt[m16 * Sz + c8 * 32 + 16 + hi * 8];
            v16h vb;
            #pragma unroll
            for (int j = 0; j < 8; ++j) { vb[j] = v0[j]; vb[j + 8] = v1[j]; }

            oacc = __builtin_amdgcn_wmma_f32_16x16x32_f16(false, pa, false, vb,
                                                          (short)0, oacc, false, false);
        }
        #pragma unroll
        for (int r = 0; r < 8; ++r) {
            int q = qbase + r + hi * 8;
            out[(size_t)(b * Sz + q) * Dz + h * DHz + m16] = oacc[r];
        }
    }
}

// =====================================================================
// GNN helpers
// =====================================================================
__global__ void deg_kernel(const int* __restrict__ ei, float* __restrict__ deg) {
    int e = blockIdx.x * blockDim.x + threadIdx.x;
    if (e < Ez) atomicAdd(&deg[ei[Ez + e]], 1.0f);
}

// one block per edge -> uniform (scalar) loads of the endpoints
__global__ void edge_agg_kernel(const int* __restrict__ ei,
                                const float* __restrict__ h,
                                float* __restrict__ agg) {
    int e = blockIdx.x;
    int f = threadIdx.x;
    int src = ei[e], dst = ei[Ez + e];
    atomicAdd(&agg[(size_t)dst * blockDim.x + f], h[(size_t)src * blockDim.x + f]);
}

__global__ void hin_kernel(const float* __restrict__ h, const float* __restrict__ agg,
                           const float* __restrict__ deg, float* __restrict__ hin, int dim) {
    int idx = blockIdx.x * blockDim.x + threadIdx.x;
    if (idx < Nn * dim) {
        int i = idx / dim;
        hin[idx] = h[idx] + agg[idx] / (deg[i] + 1.0f);
    }
}

// =====================================================================
// Fusion / assembly / LN / gather / padding
// =====================================================================
__global__ void fuse_in_kernel(const float* __restrict__ temporal,
                               const int* __restrict__ highway,
                               const float* __restrict__ ce_tab,
                               float* __restrict__ fin) {       // [BS, 32] padded
    int idx = blockIdx.x * blockDim.x + threadIdx.x;
    if (idx >= BS * FUSEKP) return;
    int t = idx / FUSEKP, f = idx - t * FUSEKP;
    float v = 0.0f;
    if (f < TDz)            v = temporal[t * TDz + f];
    else if (f < FUSEK)     v = ce_tab[highway[t] * CEDz + (f - TDz)];
    fin[idx] = v;
}

// pad fl_w [22,128] -> [32,128] (zero rows 22..31) so the fusion GEMM
// stages a fully in-bounds K=32 slab
__global__ void pad_flw_kernel(const float* __restrict__ w, float* __restrict__ wpad) {
    int idx = blockIdx.x * blockDim.x + threadIdx.x;
    if (idx < FUSEKP * Dz) {
        int r = idx / Dz, c = idx - r * Dz;
        wpad[idx] = (r < FUSEK) ? w[(size_t)r * Dz + c] : 0.0f;
    }
}

__global__ void assemble_x_kernel(const float* __restrict__ spe,
                                  const int* __restrict__ traj,
                                  const int* __restrict__ uid,
                                  const float* __restrict__ ue_tab,
                                  const float* __restrict__ fuse,
                                  float* __restrict__ x) {
    int idx = blockIdx.x * blockDim.x + threadIdx.x;
    if (idx >= BS * Dz) return;
    int t = idx / Dz, f = idx - t * Dz;
    int b = t / Sz, s = t - b * Sz;
    int i2 = f & ~1;
    float ang = (float)s * expf(-(float)i2 * (logf(10000.0f) / (float)Dz));
    float pe = (f & 1) ? cosf(ang) : sinf(ang);
    x[idx] = spe[(size_t)traj[t] * Dz + f] + ue_tab[(size_t)uid[b] * Dz + f]
           + fuse[idx] + pe;
}

__global__ __launch_bounds__(128)
void layernorm_kernel(const float* __restrict__ in, const float* __restrict__ g,
                      const float* __restrict__ bta, float* __restrict__ out) {
    __shared__ float red[128];
    int row = blockIdx.x, f = threadIdx.x;
    float v = in[(size_t)row * Dz + f];
    red[f] = v;  __syncthreads();
    for (int s = 64; s > 0; s >>= 1) { if (f < s) red[f] += red[f + s]; __syncthreads(); }
    float mean = red[0] / (float)Dz;  __syncthreads();
    float d = v - mean;
    red[f] = d * d;  __syncthreads();
    for (int s = 64; s > 0; s >>= 1) { if (f < s) red[f] += red[f + s]; __syncthreads(); }
    float var = red[0] / (float)Dz;
    out[(size_t)row * Dz + f] = d * rsqrtf(var + 1e-5f) * g[f] + bta[f];
}

__global__ void gather_last_kernel(const float* __restrict__ x,
                                   const int* __restrict__ end_idx,
                                   float* __restrict__ temb) {
    int idx = blockIdx.x * blockDim.x + threadIdx.x;
    if (idx < Bz * Dz) {
        int b = idx / Dz, f = idx - b * Dz;
        temb[idx] = x[(size_t)(b * Sz + end_idx[b]) * Dz + f];
    }
}

__global__ void pad_head_kernel(const float* __restrict__ w, const float* __restrict__ bs,
                                float* __restrict__ wpad, float* __restrict__ bpad) {
    int idx = blockIdx.x * blockDim.x + threadIdx.x;
    if (idx < Dz * UPz) {
        int r = idx / UPz, c = idx - r * UPz;
        wpad[idx] = (c < Uz) ? w[(size_t)r * Uz + c] : 0.0f;
        if (r == 0) bpad[c] = (c < Uz) ? bs[c] : 0.0f;
    }
}

__global__ void copy_out_kernel(const float* __restrict__ opad, float* __restrict__ o) {
    int idx = blockIdx.x * blockDim.x + threadIdx.x;
    if (idx < Bz * Uz) {
        int b = idx / Uz, c = idx - b * Uz;
        o[idx] = opad[(size_t)b * UPz + c];
    }
}

// =====================================================================
// launcher
// =====================================================================
static inline dim3 gemm_grid(int M, int N) { return dim3(N / 64, M / 32); }

extern "C" void kernel_launch(void* const* d_in, const int* in_sizes, int n_in,
                              void* d_out, int out_size, void* d_ws, size_t ws_size,
                              hipStream_t stream) {
    const float* node_feature = (const float*)d_in[0];
    const int*   edge_index   = (const int*)  d_in[1];
    const int*   traj_x       = (const int*)  d_in[2];
    const float* temporal_x   = (const float*)d_in[3];
    const int*   user_id_x    = (const int*)  d_in[4];
    const int*   mask         = (const int*)  d_in[5];
    const int*   end_idx      = (const int*)  d_in[6];
    const float* tmat         = (const float*)d_in[8];
    const float* dmat         = (const float*)d_in[9];
    const int*   highway_x    = (const int*)  d_in[10];
    const float* lambda2      = (const float*)d_in[11];
    const float* gnn_w1 = (const float*)d_in[12];
    const float* gnn_b1 = (const float*)d_in[13];
    const float* gnn_w2 = (const float*)d_in[14];
    const float* gnn_b2 = (const float*)d_in[15];
    const float* ue_tab = (const float*)d_in[16];
    const float* ce_tab = (const float*)d_in[17];
    const float* fl_w   = (const float*)d_in[18];
    const float* fl_b   = (const float*)d_in[19];
    const float* wqkv   = (const float*)d_in[20];
    const float* bqkv   = (const float*)d_in[21];
    const float* wo     = (const float*)d_in[22];
    const float* bo     = (const float*)d_in[23];
    const float* ln1_g  = (const float*)d_in[24];
    const float* ln1_b  = (const float*)d_in[25];
    const float* ln2_g  = (const float*)d_in[26];
    const float* ln2_b  = (const float*)d_in[27];
    const float* ffn_w1 = (const float*)d_in[28];
    const float* ffn_b1 = (const float*)d_in[29];
    const float* ffn_w2 = (const float*)d_in[30];
    const float* ffn_b2 = (const float*)d_in[31];
    const float* head_w = (const float*)d_in[32];
    const float* head_b = (const float*)d_in[33];

    // ---- workspace layout (floats) ----
    float* ws    = (float*)d_ws;
    float* spe   = ws;                               // [N,D]   persistent
    float* biasb = spe + (size_t)Nn * Dz;            // [B,S,S] persistent
    float* pool  = biasb + (size_t)Bz * Sz * Sz;

    // GNN phase (recycled)
    float* deg = pool;
    float* agg = deg + Nn;
    float* hin = agg + (size_t)Nn * Dz;
    float* h2  = hin + (size_t)Nn * Dz;

    // transformer phase (aliases GNN scratch)
    float* x     = pool;
    float* un    = x    + (size_t)BS * Dz;      // qkv [BS,384] / ffn hidden [BS,512]
    float* attn  = un   + (size_t)BS * FFz;
    float* tmp   = attn + (size_t)BS * Dz;
    float* fin   = tmp  + (size_t)BS * Dz;      // [BS,32]
    float* fout  = fin  + (size_t)BS * FUSEKP;  // [BS,128]
    float* temb  = fout + (size_t)BS * Dz;      // [64,128]
    float* wpad  = temb + (size_t)Bz * Dz;      // [128,5056]
    float* bpad  = wpad + (size_t)Dz * UPz;     // [5056]
    float* opad  = bpad + UPz;                  // [64,5056]
    float* flwp  = opad + (size_t)Bz * UPz;     // [32,128]

    const int T = 256;

    // ================= attention bias (once) =================
    bias_pre_kernel<<<(Bz * Sz * Sz + T - 1) / T, T, 0, stream>>>(
        tmat, dmat, mask, lambda2, biasb);

    // ================= GNN encoder =================
    hipMemsetAsync(deg, 0, (size_t)Nn * sizeof(float), stream);
    hipMemsetAsync(agg, 0, (size_t)Nn * FEAz * sizeof(float), stream);
    deg_kernel<<<(Ez + T - 1) / T, T, 0, stream>>>(edge_index, deg);
    edge_agg_kernel<<<Ez, FEAz, 0, stream>>>(edge_index, node_feature, agg);
    hin_kernel<<<(Nn * FEAz + T - 1) / T, T, 0, stream>>>(node_feature, agg, deg, hin, FEAz);
    gemm_wmma_f16<<<gemm_grid(Nn, Dz), T, 0, stream>>>(
        hin, gnn_w1, gnn_b1, nullptr, h2, Nn, Dz, FEAz, 1);

    hipMemsetAsync(agg, 0, (size_t)Nn * Dz * sizeof(float), stream);
    edge_agg_kernel<<<Ez, Dz, 0, stream>>>(edge_index, h2, agg);
    hin_kernel<<<(Nn * Dz + T - 1) / T, T, 0, stream>>>(h2, agg, deg, hin, Dz);
    gemm_wmma_f16<<<gemm_grid(Nn, Dz), T, 0, stream>>>(
        hin, gnn_w2, gnn_b2, nullptr, spe, Nn, Dz, Dz, 1);

    // ================= embedding fusion + x =================
    pad_flw_kernel<<<(FUSEKP * Dz + T - 1) / T, T, 0, stream>>>(fl_w, flwp);
    fuse_in_kernel<<<(BS * FUSEKP + T - 1) / T, T, 0, stream>>>(
        temporal_x, highway_x, ce_tab, fin);
    gemm_wmma_f16<<<gemm_grid(BS, Dz), T, 0, stream>>>(
        fin, flwp, fl_b, nullptr, fout, BS, Dz, FUSEKP, 1);
    assemble_x_kernel<<<(BS * Dz + T - 1) / T, T, 0, stream>>>(
        spe, traj_x, user_id_x, ue_tab, fout, x);

    // ================= transformer layers =================
    const size_t attn_lds = 2 * Sz * DHz * sizeof(_Float16)
                          + 8 * 16 * Sz * sizeof(float);
    for (int l = 0; l < Lz; ++l) {
        const float* wq  = wqkv   + (size_t)l * Dz * 3 * Dz;
        const float* bq  = bqkv   + (size_t)l * 3 * Dz;
        const float* wol = wo     + (size_t)l * Dz * Dz;
        const float* bol = bo     + (size_t)l * Dz;
        const float* w1  = ffn_w1 + (size_t)l * Dz * FFz;
        const float* b1  = ffn_b1 + (size_t)l * FFz;
        const float* w2  = ffn_w2 + (size_t)l * FFz * Dz;
        const float* b2  = ffn_b2 + (size_t)l * Dz;

        gemm_wmma_f16<<<gemm_grid(BS, 3 * Dz), T, 0, stream>>>(
            x, wq, bq, nullptr, un, BS, 3 * Dz, Dz, 0);
        attn_wmma<<<Bz * Hz, T, attn_lds, stream>>>(un, biasb, attn);
        gemm_wmma_f16<<<gemm_grid(BS, Dz), T, 0, stream>>>(
            attn, wol, bol, x, tmp, BS, Dz, Dz, 0);
        layernorm_kernel<<<BS, 128, 0, stream>>>(
            tmp, ln1_g + (size_t)l * Dz, ln1_b + (size_t)l * Dz, x);
        gemm_wmma_f16<<<gemm_grid(BS, FFz), T, 0, stream>>>(
            x, w1, b1, nullptr, un, BS, FFz, Dz, 1);
        gemm_wmma_f16<<<gemm_grid(BS, Dz), T, 0, stream>>>(
            un, w2, b2, x, tmp, BS, Dz, FFz, 0);
        layernorm_kernel<<<BS, 128, 0, stream>>>(
            tmp, ln2_g + (size_t)l * Dz, ln2_b + (size_t)l * Dz, x);
    }

    // ================= head (padded to N=5056) =================
    pad_head_kernel<<<(Dz * UPz + T - 1) / T, T, 0, stream>>>(head_w, head_b, wpad, bpad);
    gather_last_kernel<<<(Bz * Dz + T - 1) / T, T, 0, stream>>>(x, end_idx, temb);
    gemm_wmma_f16<<<gemm_grid(Bz, UPz), T, 0, stream>>>(
        temb, wpad, bpad, nullptr, opad, Bz, UPz, Dz, 0);
    copy_out_kernel<<<(Bz * Uz + T - 1) / T, T, 0, stream>>>(opad, (float*)d_out);
}
